// UResNet_80453327389031
// MI455X (gfx1250) — compile-verified, matrix operand order checked by reading, not run on hardware
//
#include <hip/hip_runtime.h>
#include <cstdint>
#include <cstddef>

#define BN_EPS 1e-4f
#define LEAKY  0.25f

typedef __bf16 v16bf         __attribute__((ext_vector_type(16)));
typedef float  v8f           __attribute__((ext_vector_type(8)));
typedef unsigned short u16x8 __attribute__((ext_vector_type(8)));
typedef unsigned short u16x16 __attribute__((ext_vector_type(16)));
typedef int i32x4            __attribute__((ext_vector_type(4)));

#if __has_builtin(__builtin_amdgcn_global_load_async_to_lds_b128)
#define HAVE_ASYNC_LDS 1
typedef __attribute__((address_space(1))) i32x4 g_i32x4;   // global (cuda_device)
typedef __attribute__((address_space(3))) i32x4 l_i32x4;   // LDS
#endif

__device__ __forceinline__ uint16_t f2bf_bits(float f) {
  union { float f; uint32_t u; } v; v.f = f;
  uint32_t r = v.u + 0x7FFFu + ((v.u >> 16) & 1u);   // round-to-nearest-even
  return (uint16_t)(r >> 16);
}

// ---------------------------------------------------------------------------
// One-time weight transform: W (27,a,b) fp32 -> Wt (27, b, a+8) bf16,
// transposed [k][col][cin] with +8 pad so the conv kernel's LDS stage is a
// pure 16B-aligned copy and B fragments are two ds_load_b128.
// ---------------------------------------------------------------------------
__global__ void wtrans_kernel(const float* __restrict__ W,
                              unsigned short* __restrict__ Wt, int a, int b) {
  long t = (long)blockIdx.x * blockDim.x + threadIdx.x;
  long total = 27L * a * b;
  if (t >= total) return;
  int k  = (int)(t / (a * b));
  int r  = (int)(t - (long)k * a * b);
  int cc = r / a;
  int ci = r - cc * a;
  Wt[(size_t)k * b * (a + 8) + (size_t)cc * (a + 8) + ci] =
      f2bf_bits(W[(size_t)k * a * b + (size_t)ci * b + cc]);
}

// ---------------------------------------------------------------------------
// Submanifold 3^3 conv as gather-GEMM on bf16 WMMA.
//   x   : bf16 activations (n x a), a % 32 == 0
//   Wt  : bf16 transposed weights (27, b, a+8)
//   out : fp32 (n x b), b == NT*16
// Block = 256 threads = 8 waves. Wave w owns rows [bx*128+16w, +16) and ALL
// NT column tiles (A gathered once, reused NT times). Per-k weight tile is
// copied to LDS via async global->LDS (ASYNCcnt) when available.
// ---------------------------------------------------------------------------
template <int NT>
__global__ __launch_bounds__(256) void subconv_wmma_kernel(
    const unsigned short* __restrict__ x, const int* __restrict__ nbr,
    const unsigned short* __restrict__ Wt, float* __restrict__ out,
    int n, int a)
{
  extern __shared__ unsigned short sw[];   // [NT*16][a+8] bf16 bits
  const int b    = NT * 16;
  const int AP   = a + 8;
  const int tile = b * AP;                 // elements per k-slice
  const int tid  = threadIdx.x;
  const int wave = tid >> 5;
  const int lane = tid & 31;
  const int m    = lane & 15;     // A: row-in-tile, B/D: column-in-tile
  const int half = lane >> 4;
  const int rowbase = blockIdx.x * 128 + wave * 16;
  const int arow    = rowbase + m;
  const bool rowOK  = arow < n;

  v8f acc[NT] = {};

  for (int k = 0; k < 27; ++k) {
    // stage Wt[k] -> LDS: pure 16-byte copies (async to LDS when available)
    const unsigned short* wk = Wt + (size_t)k * tile;
#if defined(HAVE_ASYNC_LDS)
    for (int e = tid * 8; e < tile; e += 2048) {
      __builtin_amdgcn_global_load_async_to_lds_b128(
          (g_i32x4*)(wk + e), (l_i32x4*)(sw + e), 0, 0);
    }
#if __has_builtin(__builtin_amdgcn_s_wait_asynccnt)
    __builtin_amdgcn_s_wait_asynccnt(0);
#else
    asm volatile("s_wait_asynccnt 0" ::: "memory");
#endif
#else
    for (int e = tid * 8; e < tile; e += 2048)
      *(u16x8*)(sw + e) = *(const u16x8*)(wk + e);
#endif
    __syncthreads();

    const int src = rowOK ? nbr[(size_t)arow * 27 + k] : -1;
    const unsigned short* xr = (src >= 0) ? (x + (size_t)src * a) : nullptr;

    for (int kc = 0; kc < a; kc += 32) {
      // A fragment: K = kc + 8*half + [0..7]  and  kc + 16 + 8*half + [0..7]
      u16x8 alo = {0, 0, 0, 0, 0, 0, 0, 0};
      u16x8 ahi = {0, 0, 0, 0, 0, 0, 0, 0};
      if (xr) {
        alo = *(const u16x8*)(xr + kc + 8 * half);
        ahi = *(const u16x8*)(xr + kc + 16 + 8 * half);
      }
      const u16x16 araw = __builtin_shufflevector(
          alo, ahi, 0, 1, 2, 3, 4, 5, 6, 7, 8, 9, 10, 11, 12, 13, 14, 15);
      const v16bf A = __builtin_bit_cast(v16bf, araw);

#pragma unroll
      for (int ct = 0; ct < NT; ++ct) {
        const unsigned short* bp = sw + (size_t)(ct * 16 + m) * AP + kc;
        const u16x8 blo = *(const u16x8*)(bp + 8 * half);
        const u16x8 bhi = *(const u16x8*)(bp + 16 + 8 * half);
        const u16x16 braw = __builtin_shufflevector(
            blo, bhi, 0, 1, 2, 3, 4, 5, 6, 7, 8, 9, 10, 11, 12, 13, 14, 15);
        const v16bf B = __builtin_bit_cast(v16bf, braw);
        acc[ct] = __builtin_amdgcn_wmma_f32_16x16x32_bf16(
            false, A, false, B, (short)0, acc[ct], false, false);
      }
    }
    __syncthreads();
  }

#pragma unroll
  for (int ct = 0; ct < NT; ++ct) {
#pragma unroll
    for (int r = 0; r < 8; ++r) {        // D layout: VGPR r -> M = r + 8*half
      int orow = rowbase + r + 8 * half;
      if (orow < n) out[(size_t)orow * b + ct * 16 + m] = acc[ct][r];
    }
  }
}

// ---------------------------------------------------------------------------
// Initial submanifold conv, a=1 (features) -> 32 channels: trivial VALU kernel.
// ---------------------------------------------------------------------------
__global__ void conv_in_kernel(const float* __restrict__ f,
                               const int* __restrict__ nbr,
                               const float* __restrict__ W,  // (27,1,32)
                               float* __restrict__ out, int n) {
  long t = (long)blockIdx.x * blockDim.x + threadIdx.x;
  if (t >= (long)n * 32) return;
  int i = (int)(t >> 5), c = (int)(t & 31);
  float s = 0.f;
#pragma unroll
  for (int k = 0; k < 27; ++k) {
    int s0 = nbr[(size_t)i * 27 + k];
    if (s0 >= 0) s += f[s0] * W[k * 32 + c];
  }
  out[t] = s;
}

// ---------------------------------------------------------------------------
// BN(Leaky)ReLU over active sites: per-channel mean/var then normalize.
// ---------------------------------------------------------------------------
__global__ void bn_reduce_kernel(const float* __restrict__ x, int n, int C,
                                 float* __restrict__ stats) {
  __shared__ float ss[256], sq[256];
  const int c = blockIdx.x, tid = threadIdx.x;
  float s = 0.f, q = 0.f;
  for (int i = tid; i < n; i += 256) {
    float v = x[(size_t)i * C + c];
    s += v; q += v * v;
  }
  ss[tid] = s; sq[tid] = q;
  __syncthreads();
  for (int o = 128; o > 0; o >>= 1) {
    if (tid < o) { ss[tid] += ss[tid + o]; sq[tid] += sq[tid + o]; }
    __syncthreads();
  }
  if (tid == 0) {
    float mu = ss[0] / (float)n;
    stats[2 * c]     = mu;
    stats[2 * c + 1] = sq[0] / (float)n - mu * mu;
  }
}

__global__ void bn_apply_kernel(const float* __restrict__ x,
                                const float* __restrict__ stats,
                                const float* __restrict__ g,
                                const float* __restrict__ b,
                                float* __restrict__ y,
                                long total, int C, float leak) {
  long i = (long)blockIdx.x * blockDim.x + threadIdx.x;
  if (i >= total) return;
  int c = (int)(i % C);
  float mu = stats[2 * c], var = stats[2 * c + 1];
  float v = (x[i] - mu) * rsqrtf(var + BN_EPS) * g[c] + b[c];
  y[i] = v > 0.f ? v : leak * v;
}

// variant writing bf16 bits (feeds the WMMA conv: halves gather traffic)
__global__ void bn_apply_bf16_kernel(const float* __restrict__ x,
                                     const float* __restrict__ stats,
                                     const float* __restrict__ g,
                                     const float* __restrict__ b,
                                     unsigned short* __restrict__ y,
                                     long total, int C, float leak) {
  long i = (long)blockIdx.x * blockDim.x + threadIdx.x;
  if (i >= total) return;
  int c = (int)(i % C);
  float mu = stats[2 * c], var = stats[2 * c + 1];
  float v = (x[i] - mu) * rsqrtf(var + BN_EPS) * g[c] + b[c];
  v = v > 0.f ? v : leak * v;
  y[i] = f2bf_bits(v);
}

// ---------------------------------------------------------------------------
// Strided size-2 down conv (scatter into coarse parents) and transposed conv.
// ---------------------------------------------------------------------------
__global__ void downconv_kernel(const float* __restrict__ x,
                                const int* __restrict__ parent,
                                const int* __restrict__ kid,
                                const float* __restrict__ W,
                                float* __restrict__ out, int n, int a, int q) {
  long t = (long)blockIdx.x * blockDim.x + threadIdx.x;
  long total = (long)n * q;
  if (t >= total) return;
  int i = (int)(t / q), c = (int)(t % q);
  const float* w  = W + ((size_t)kid[i] * a) * q + c;
  const float* xr = x + (size_t)i * a;
  float s = 0.f;
  for (int j = 0; j < a; ++j) s += xr[j] * w[(size_t)j * q];
  atomicAdd(&out[(size_t)parent[i] * q + c], s);
}

__global__ void deconv_kernel(const float* __restrict__ y,
                              const int* __restrict__ parent,
                              const int* __restrict__ kid,
                              const float* __restrict__ W,
                              float* __restrict__ out, int n, int q, int p) {
  long t = (long)blockIdx.x * blockDim.x + threadIdx.x;
  long total = (long)n * p;
  if (t >= total) return;
  int i = (int)(t / p), c = (int)(t % p);
  const float* g = y + (size_t)parent[i] * q;
  const float* w = W + ((size_t)kid[i] * q) * p + c;
  float s = 0.f;
  for (int j = 0; j < q; ++j) s += g[j] * w[(size_t)j * p];
  out[t] = s;
}

// ---------------------------------------------------------------------------
// Elementwise helpers
// ---------------------------------------------------------------------------
__global__ void matmul_bias_kernel(const float* __restrict__ x,
                                   const float* __restrict__ W,
                                   const float* __restrict__ bias,
                                   float* __restrict__ y, int n, int a, int b) {
  long t = (long)blockIdx.x * blockDim.x + threadIdx.x;
  long total = (long)n * b;
  if (t >= total) return;
  int i = (int)(t / b), c = (int)(t % b);
  const float* xr = x + (size_t)i * a;
  float s = bias ? bias[c] : 0.f;
  for (int j = 0; j < a; ++j) s += xr[j] * W[(size_t)j * b + c];
  y[t] = s;
}

__global__ void add2_kernel(float* __restrict__ dst, const float* __restrict__ s1,
                            const float* __restrict__ s2, long total) {
  long i = (long)blockIdx.x * blockDim.x + threadIdx.x;
  if (i < total) dst[i] = s1[i] + s2[i];
}

__global__ void concat2_kernel(const float* __restrict__ x, const float* __restrict__ y,
                               float* __restrict__ cat, int n, int p) {
  long t = (long)blockIdx.x * blockDim.x + threadIdx.x;
  long total = (long)n * 2 * p;
  if (t >= total) return;
  int i = (int)(t / (2 * p)), c = (int)(t % (2 * p));
  cat[t] = (c < p) ? x[(size_t)i * p + c] : y[(size_t)i * p + (c - p)];
}

__global__ void zero_kernel(float* __restrict__ p, long total) {
  long i = (long)blockIdx.x * blockDim.x + threadIdx.x;
  if (i < total) p[i] = 0.f;
}

// ---------------------------------------------------------------------------
// Host orchestration
// ---------------------------------------------------------------------------
struct RBp {
  const float *W1, *W2, *Wsc, *b1, *b2, *g1, *g2;
  unsigned short *W1t, *W2t;   // bf16 transposed weights (workspace)
};
struct LVp { const float *Wd, *Wu, *bd, *bu, *gd, *gu; RBp post[2]; RBp pre[2]; };
struct Ctx {
  hipStream_t stream;
  const float* features;
  const float *W_in, *Wlin, *bfp, *blin, *gfp;
  LVp lv[5];
  const int *nbr[5], *parent[4], *kid[4];
  int n[5];
  float *x[5], *t0, *t1, *cat, *stats;
  unsigned short* tb;   // bf16 staging for conv inputs
};

static inline int cdiv(long a, long b) { return (int)((a + b - 1) / b); }

static void bnact(Ctx& c, const float* x, const float* g, const float* b,
                  float* y, int n, int C, float leak) {
  bn_reduce_kernel<<<C, 256, 0, c.stream>>>(x, n, C, c.stats);
  long total = (long)n * C;
  bn_apply_kernel<<<cdiv(total, 256), 256, 0, c.stream>>>(x, c.stats, g, b, y, total, C, leak);
}

static void bnact_bf16(Ctx& c, const float* x, const float* g, const float* b,
                       unsigned short* y, int n, int C, float leak) {
  bn_reduce_kernel<<<C, 256, 0, c.stream>>>(x, n, C, c.stats);
  long total = (long)n * C;
  bn_apply_bf16_kernel<<<cdiv(total, 256), 256, 0, c.stream>>>(x, c.stats, g, b, y, total, C, leak);
}

static void subconv(Ctx& c, const unsigned short* x, const int* nbr,
                    const unsigned short* Wt, float* out, int n, int a, int b) {
  dim3 grid((unsigned)cdiv(n, 128));
  size_t sh = (size_t)b * (size_t)(a + 8) * sizeof(unsigned short);
  switch (b / 16) {
    case 2:  subconv_wmma_kernel<2> <<<grid, 256, sh, c.stream>>>(x, nbr, Wt, out, n, a); break;
    case 4:  subconv_wmma_kernel<4> <<<grid, 256, sh, c.stream>>>(x, nbr, Wt, out, n, a); break;
    case 6:  subconv_wmma_kernel<6> <<<grid, 256, sh, c.stream>>>(x, nbr, Wt, out, n, a); break;
    case 8:  subconv_wmma_kernel<8> <<<grid, 256, sh, c.stream>>>(x, nbr, Wt, out, n, a); break;
    case 10: subconv_wmma_kernel<10><<<grid, 256, sh, c.stream>>>(x, nbr, Wt, out, n, a); break;
    default: break;
  }
}

static void resblock(Ctx& c, const float* xin, float* xout, const RBp& rb,
                     const int* nbr, int n, int a, int b) {
  bnact_bf16(c, xin, rb.g1, rb.b1, c.tb, n, a, LEAKY);
  subconv(c, c.tb, nbr, rb.W1t, c.t1, n, a, b);
  bnact_bf16(c, c.t1, rb.g2, rb.b2, c.tb, n, b, LEAKY);
  subconv(c, c.tb, nbr, rb.W2t, c.t1, n, b, b);
  long total = (long)n * b;
  if (rb.Wsc) {  // NetworkInNetwork shortcut
    matmul_bias_kernel<<<cdiv(total, 256), 256, 0, c.stream>>>(xin, rb.Wsc, nullptr, c.t0, n, a, b);
    add2_kernel<<<cdiv(total, 256), 256, 0, c.stream>>>(xout, c.t0, c.t1, total);
  } else {
    add2_kernel<<<cdiv(total, 256), 256, 0, c.stream>>>(xout, xin, c.t1, total);
  }
}

static void run_level(Ctx& c, int l) {
  const int n = c.n[l], p = 32 * (l + 1);
  float* x = c.x[l];
  for (int r = 0; r < 2; ++r) resblock(c, x, x, c.lv[l].pre[r], c.nbr[l], n, p, p);
  if (l < 4) {
    const int q = 32 * (l + 2), n2 = c.n[l + 1];
    bnact(c, x, c.lv[l].gd, c.lv[l].bd, c.t0, n, p, LEAKY);
    long zc = (long)n2 * q;
    zero_kernel<<<cdiv(zc, 256), 256, 0, c.stream>>>(c.x[l + 1], zc);
    long td = (long)n * q;
    downconv_kernel<<<cdiv(td, 256), 256, 0, c.stream>>>(
        c.t0, c.parent[l], c.kid[l], c.lv[l].Wd, c.x[l + 1], n, p, q);
    run_level(c, l + 1);
    bnact(c, c.x[l + 1], c.lv[l].gu, c.lv[l].bu, c.t0, n2, q, LEAKY);
    long tu = (long)n * p;
    deconv_kernel<<<cdiv(tu, 256), 256, 0, c.stream>>>(
        c.t0, c.parent[l], c.kid[l], c.lv[l].Wu, c.t1, n, q, p);
    long tc = (long)n * 2 * p;
    concat2_kernel<<<cdiv(tc, 256), 256, 0, c.stream>>>(x, c.t1, c.cat, n, p);
    resblock(c, c.cat, x, c.lv[l].post[0], c.nbr[l], n, 2 * p, p);
    resblock(c, x, x, c.lv[l].post[1], c.nbr[l], n, p, p);
  }
}

extern "C" void kernel_launch(void* const* d_in, const int* in_sizes, int n_in,
                              void* d_out, int out_size, void* d_ws, size_t ws_size,
                              hipStream_t stream) {
  (void)n_in; (void)out_size; (void)ws_size;
  Ctx c{};
  c.stream = stream;
  int idx = 0;
  auto F = [&](int i) { return (const float*)d_in[i]; };
  auto I = [&](int i) { return (const int*)d_in[i]; };

  // pytree order: features; params (sorted keys); rulebooks (tuples in order)
  c.features = F(idx++);
  c.W_in = F(idx++); c.Wlin = F(idx++); c.bfp = F(idx++);
  c.blin = F(idx++); c.gfp = F(idx++);
  auto parse_rb = [&](bool sc) {
    RBp r{};
    r.W1 = F(idx++); r.W2 = F(idx++);
    r.Wsc = sc ? F(idx++) : nullptr;
    r.b1 = F(idx++); r.b2 = F(idx++); r.g1 = F(idx++); r.g2 = F(idx++);
    return r;
  };
  for (int l = 0; l < 5; ++l) {
    if (l < 4) {  // sorted: Wd,Wu,bd,bu,gd,gu,post,pre
      c.lv[l].Wd = F(idx++); c.lv[l].Wu = F(idx++);
      c.lv[l].bd = F(idx++); c.lv[l].bu = F(idx++);
      c.lv[l].gd = F(idx++); c.lv[l].gu = F(idx++);
      c.lv[l].post[0] = parse_rb(true);
      c.lv[l].post[1] = parse_rb(false);
    }
    c.lv[l].pre[0] = parse_rb(false);
    c.lv[l].pre[1] = parse_rb(false);
  }
  for (int l = 0; l < 5; ++l) { c.nbr[l] = I(idx); c.n[l] = in_sizes[idx] / 27; idx++; }
  for (int l = 0; l < 4; ++l) c.parent[l] = I(idx++);
  for (int l = 0; l < 4; ++l) c.kid[l] = I(idx++);

  // bump-allocate workspace
  char* base = (char*)d_ws; size_t off = 0;
  auto alloc_b = [&](size_t bytes) -> void* {
    void* p = (void*)(base + off);
    off += (bytes + 255) / 256 * 256;
    return p;
  };
  c.stats = (float*)alloc_b(1024 * sizeof(float));
  size_t smax = 0;
  for (int l = 0; l < 5; ++l) {
    size_t s = (size_t)c.n[l] * 2 * (size_t)(32 * (l + 1));
    if (s > smax) smax = s;
  }
  c.t0  = (float*)alloc_b(smax * sizeof(float));
  c.t1  = (float*)alloc_b(smax * sizeof(float));
  c.cat = (float*)alloc_b(smax * sizeof(float));
  c.tb  = (unsigned short*)alloc_b(smax * sizeof(unsigned short));
  for (int l = 0; l < 5; ++l)
    c.x[l] = (float*)alloc_b((size_t)c.n[l] * (size_t)(32 * (l + 1)) * sizeof(float));

  // one-time (per launch) weight transform: fp32 -> transposed padded bf16
  auto prep_w = [&](const float* W, int a, int b) -> unsigned short* {
    size_t cnt = 27ULL * (size_t)b * (size_t)(a + 8);
    unsigned short* Wt = (unsigned short*)alloc_b(cnt * sizeof(unsigned short));
    long total = 27L * a * b;
    wtrans_kernel<<<cdiv(total, 256), 256, 0, stream>>>(W, Wt, a, b);
    return Wt;
  };
  for (int l = 0; l < 5; ++l) {
    const int p = 32 * (l + 1);
    c.lv[l].pre[0].W1t = prep_w(c.lv[l].pre[0].W1, p, p);
    c.lv[l].pre[0].W2t = prep_w(c.lv[l].pre[0].W2, p, p);
    c.lv[l].pre[1].W1t = prep_w(c.lv[l].pre[1].W1, p, p);
    c.lv[l].pre[1].W2t = prep_w(c.lv[l].pre[1].W2, p, p);
    if (l < 4) {
      c.lv[l].post[0].W1t = prep_w(c.lv[l].post[0].W1, 2 * p, p);
      c.lv[l].post[0].W2t = prep_w(c.lv[l].post[0].W2, p, p);
      c.lv[l].post[1].W1t = prep_w(c.lv[l].post[1].W1, p, p);
      c.lv[l].post[1].W2t = prep_w(c.lv[l].post[1].W2, p, p);
    }
  }

  const int n0 = c.n[0];
  // initial submanifold conv (1 -> 32 channels), trivial VALU kernel
  long t0n = (long)n0 * 32;
  conv_in_kernel<<<cdiv(t0n, 256), 256, 0, stream>>>(c.features, c.nbr[0], c.W_in, c.x[0], n0);
  run_level(c, 0);
  // final BN-ReLU (leak=0) + linear head to 5 classes
  bnact(c, c.x[0], c.gfp, c.bfp, c.t0, n0, 32, 0.0f);
  long tf = (long)n0 * 5;
  matmul_bias_kernel<<<cdiv(tf, 256), 256, 0, stream>>>(
      c.t0, c.Wlin, c.blin, (float*)d_out, n0, 32, 5);
}
